// RWKV_ATTN_90623809946181
// MI455X (gfx1250) — compile-verified, matrix operand order checked by reading, not run on hardware
//
#include <hip/hip_runtime.h>
#include <hip/hip_bf16.h>
#include <cstdint>
#include <cstddef>

// ---------------- problem constants ----------------
#define BATCH 2
#define NPTS  2048
#define INCH  128
#define DIM   256
#define KNN   16
#define POSH  64
#define FFNH  1024
#define PTS   (BATCH * NPTS)
#define EPSLN 1e-5f

typedef __bf16 bf16_t;
typedef __attribute__((ext_vector_type(8)))  __bf16 v8bf;
typedef __attribute__((ext_vector_type(16))) __bf16 v16bf;
typedef __attribute__((ext_vector_type(8)))  float  v8f;

// ---------------- input leaf map (jax sorted-key tree flatten) ----------------
enum {
  IN_KEY = 0,
  IN_ATTN1_W = 1, IN_ATTN2_W = 2,
  IN_AMLP_B1 = 3, IN_AMLP_B2 = 4, IN_AMLP_BN_B = 5, IN_AMLP_BN_G = 6,
  IN_AMLP_BN_M = 7, IN_AMLP_BN_V = 8, IN_AMLP_W1 = 9, IN_AMLP_W2 = 10,
  IN_CEND_B = 11, IN_CEND_W = 12,
  IN_CKEY_B = 13, IN_CKEY_W = 14,
  IN_CQ_B = 15, IN_CQ_W = 16,
  IN_CV_B = 17, IN_CV_W = 18,
  IN_MLPV_C1_B = 19, IN_MLPV_C1_W = 20, IN_MLPV_C2_B = 21, IN_MLPV_C2_W = 22,
  IN_MLPV_SC_B = 23, IN_MLPV_SC_W = 24,
  IN_PMLP_B1 = 25, IN_PMLP_B2 = 26, IN_PMLP_BN_B = 27, IN_PMLP_BN_G = 28,
  IN_PMLP_BN_M = 29, IN_PMLP_BN_V = 30, IN_PMLP_W1 = 31, IN_PMLP_W2 = 32,
  IN_RWKV0 = 33, IN_RWKV1 = 53,
  IN_POS = 73, IN_QUERY = 74,
  IN_TOTAL = 75
};
// rwkv sub-leaf offsets (sorted keys)
enum {
  RW_FMIX_K = 0, RW_FMIX_R = 1, RW_FWK = 2, RW_FWR = 3, RW_FWV = 4,
  RW_LN0_B = 5, RW_LN0_G = 6, RW_LN1_B = 7, RW_LN1_G = 8, RW_LN2_B = 9, RW_LN2_G = 10,
  RW_MIX_K = 11, RW_MIX_R = 12, RW_MIX_V = 13, RW_TD = 14, RW_TF = 15,
  RW_WK = 16, RW_WO = 17, RW_WR = 18, RW_WV = 19
};

// ---------------- WMMA helpers ----------------
__device__ __forceinline__ v8f wmma_bf16(v16bf a, v16bf b, v8f c) {
  return __builtin_amdgcn_wmma_f32_16x16x32_bf16(false, a, false, b, (short)0, c,
                                                 false, false);
}

// A/B fragment load per ISA 7.12.2: lane L (m = L&15, half = L>>4) holds
// k = half*8 .. half*8+7 (elems 0..7) and k = 16+half*8 .. +7 (elems 8..15),
// i.e. two contiguous 16B (b128) loads per lane.
__device__ __forceinline__ v16bf load_frag(const bf16_t* __restrict__ base,
                                           int ld, int row, int k0, int half) {
  union { v16bf v; v8bf h[2]; } f;
  const bf16_t* p = base + (size_t)row * ld + k0 + half * 8;
  f.h[0] = *(const v8bf*)(p);
  f.h[1] = *(const v8bf*)(p + 16);
  return f.v;
}

// ---------------- generic GEMM: Out[PTS x O] = epi(X[PTS x I]b16 @ W[O x I]b16^T + bias) + Res
// one wave per 16x16 output tile. epi: 0 none, 1 relu, 2 sigmoid, 3 relu^2
__global__ __launch_bounds__(32)
void k_gemm(const bf16_t* __restrict__ X, int ldx,
            const bf16_t* __restrict__ W,
            const float* __restrict__ bias,
            const float* __restrict__ Res,
            float* __restrict__ Out,
            bf16_t* __restrict__ OutBf,
            int I, int O, int epi) {
  const int lane = threadIdx.x;
  const int m16 = lane & 15, half = lane >> 4;
  const int arow = blockIdx.y * 16 + m16;   // activation row
  const int wrow = blockIdx.x * 16 + m16;   // weight row (output channel)
  v8f acc = {};
  for (int k0 = 0; k0 < I; k0 += 32) {
    v16bf a = load_frag(X, ldx, arow, k0, half);
    v16bf b = load_frag(W, I, wrow, k0, half);
    acc = wmma_bf16(a, b, acc);
  }
  const int nc = blockIdx.x * 16 + m16;
  const float bv = bias ? bias[nc] : 0.f;
#pragma unroll
  for (int r = 0; r < 8; ++r) {
    const int m = blockIdx.y * 16 + half * 8 + r;
    float v = acc[r] + bv;
    if (epi == 1) v = fmaxf(v, 0.f);
    else if (epi == 2) v = 1.f / (1.f + __expf(-v));
    else if (epi == 3) { v = fmaxf(v, 0.f); v = v * v; }
    const size_t off = (size_t)m * O + nc;
    if (Res) v += Res[off];
    if (Out) Out[off] = v;
    if (OutBf) OutBf[off] = (bf16_t)v;
  }
}

// ---------------- small utility kernels ----------------
__global__ void k_cast_bf(const float* __restrict__ a, bf16_t* __restrict__ o, int n) {
  int e = blockIdx.x * blockDim.x + threadIdx.x;
  if (e < n) o[e] = (bf16_t)a[e];
}

// pack (B,C,N) key/query into point-major [PTS x 2*INCH] bf16 (key cols 0..127)
__global__ void k_pack_cat(const float* __restrict__ key, const float* __restrict__ query,
                           bf16_t* __restrict__ cat) {
  int e = blockIdx.x * blockDim.x + threadIdx.x;
  if (e >= PTS * 2 * INCH) return;
  int p = e >> 8, c = e & 255;
  int b = p / NPTS, n = p - b * NPTS;
  float v = (c < INCH) ? key[((size_t)b * INCH + c) * NPTS + n]
                       : query[((size_t)b * INCH + (c - INCH)) * NPTS + n];
  cat[e] = (bf16_t)v;
}

__global__ void k_bnprep(const float* __restrict__ g, const float* __restrict__ b,
                         const float* __restrict__ m, const float* __restrict__ v,
                         const float* __restrict__ convb,
                         float* __restrict__ scale, float* __restrict__ shift, int H) {
  int h = blockIdx.x * blockDim.x + threadIdx.x;
  if (h >= H) return;
  float s = g[h] * rsqrtf(v[h] + EPSLN);
  scale[h] = s;
  shift[h] = (convb[h] - m[h]) * s + b[h];
}

// LayerNorm over DIM=256 channels, one wave per token
__global__ __launch_bounds__(32)
void k_layernorm(const float* __restrict__ x, const float* __restrict__ g,
                 const float* __restrict__ bb, float* __restrict__ y) {
  int p = blockIdx.x, lane = threadIdx.x;
  const float* row = x + (size_t)p * DIM;
  float v[8];
  float s = 0.f;
#pragma unroll
  for (int j = 0; j < 8; ++j) { v[j] = row[j * 32 + lane]; s += v[j]; }
#pragma unroll
  for (int o = 16; o > 0; o >>= 1) s += __shfl_xor(s, o, 32);
  float mean = s * (1.f / DIM);
  float q = 0.f;
#pragma unroll
  for (int j = 0; j < 8; ++j) { float d = v[j] - mean; q += d * d; }
#pragma unroll
  for (int o = 16; o > 0; o >>= 1) q += __shfl_xor(q, o, 32);
  float inv = rsqrtf(q * (1.f / DIM) + EPSLN);
#pragma unroll
  for (int j = 0; j < 8; ++j) {
    int c = j * 32 + lane;
    y[(size_t)p * DIM + c] = (v[j] - mean) * inv * g[c] + bb[c];
  }
}

// token-shift mix: out = xn*mix + shift(xn)*(1-mix); any of xk/xv/xr may be null
__global__ void k_mixshift(const float* __restrict__ xn,
                           const float* __restrict__ mk, const float* __restrict__ mv,
                           const float* __restrict__ mr,
                           bf16_t* __restrict__ xk, bf16_t* __restrict__ xv,
                           bf16_t* __restrict__ xr) {
  int e = blockIdx.x * blockDim.x + threadIdx.x;
  if (e >= PTS * DIM) return;
  int p = e >> 8, c = e & 255;
  int n = p % NPTS;
  float cur = xn[e];
  float prev = (n == 0) ? 0.f : xn[e - DIM];
  if (xk) { float m = mk[c]; xk[e] = (bf16_t)(cur * m + prev * (1.f - m)); }
  if (xv) { float m = mv[c]; xv[e] = (bf16_t)(cur * m + prev * (1.f - m)); }
  if (xr) { float m = mr[c]; xr[e] = (bf16_t)(cur * m + prev * (1.f - m)); }
}

// numerically-stable WKV scan, one thread per (batch, channel)
__global__ void k_wkv(const float* __restrict__ td, const float* __restrict__ tf,
                      const float* __restrict__ k, const float* __restrict__ v,
                      float* __restrict__ out) {
  int t = blockIdx.x * blockDim.x + threadIdx.x;
  if (t >= BATCH * DIM) return;
  int b = t / DIM, c = t - b * DIM;
  float w = -__expf(td[c]);
  float u = tf[c];
  float a = 0.f, bb = 0.f, pp = -1e38f;
  for (int n = 0; n < NPTS; ++n) {
    size_t e = ((size_t)(b * NPTS + n)) * DIM + c;
    float kt = k[e], vt = v[e];
    float ww = u + kt;
    float pm = fmaxf(pp, ww);
    float e1 = __expf(pp - pm), e2 = __expf(ww - pm);
    out[e] = (e1 * a + e2 * vt) / (e1 * bb + e2);
    float ww2 = pp + w;
    float p2 = fmaxf(ww2, kt);
    float f1 = __expf(ww2 - p2), f2 = __expf(kt - p2);
    a = f1 * a + f2 * vt;
    bb = f1 * bb + f2;
    pp = p2;
  }
}

__global__ void k_mul_bf(const float* __restrict__ a, const float* __restrict__ b,
                         bf16_t* __restrict__ o, int n) {
  int e = blockIdx.x * blockDim.x + threadIdx.x;
  if (e < n) o[e] = (bf16_t)(a[e] * b[e]);
}

// o = x + r*t  (+ optional bf16 copy)
__global__ void k_fma_ew(const float* __restrict__ x, const float* __restrict__ r,
                         const float* __restrict__ t, float* __restrict__ o,
                         bf16_t* __restrict__ obf, int n) {
  int e = blockIdx.x * blockDim.x + threadIdx.x;
  if (e >= n) return;
  float v = x[e] + r[e] * t[e];
  o[e] = v;
  if (obf) obf[e] = (bf16_t)v;
}

// brute-force 16-NN per query point (per-batch), insertion list
__global__ void k_knn(const float* __restrict__ pos, int* __restrict__ idx) {
  int g = blockIdx.x * blockDim.x + threadIdx.x;
  if (g >= PTS) return;
  int b = g / NPTS, n = g - b * NPTS;
  const float* px = pos + (size_t)(b * 3 + 0) * NPTS;
  const float* py = pos + (size_t)(b * 3 + 1) * NPTS;
  const float* pz = pos + (size_t)(b * 3 + 2) * NPTS;
  float qx = px[n], qy = py[n], qz = pz[n];
  float bd[KNN];
  int bi[KNN];
#pragma unroll
  for (int i = 0; i < KNN; ++i) { bd[i] = 3.0e38f; bi[i] = 0; }
  for (int m = 0; m < NPTS; ++m) {
    float dx = qx - px[m], dy = qy - py[m], dz = qz - pz[m];
    float d = dx * dx + dy * dy + dz * dz;
    if (d < bd[KNN - 1]) {
      int j = KNN - 1;
      while (j > 0 && bd[j - 1] > d) { bd[j] = bd[j - 1]; bi[j] = bi[j - 1]; --j; }
      bd[j] = d;
      bi[j] = m;
    }
  }
#pragma unroll
  for (int i = 0; i < KNN; ++i) idx[(size_t)g * KNN + i] = bi[i];
}

// ---------------- fused per-point attention kernel ----------------
// One block (8 waves) per point: pos-MLP -> (qk_rel + pos_emb) -> attn MLP
// (256->1024->256 via WMMA from LDS) -> softmax over K -> aggregate value_g.
// Weights stream from L2 (w1+w2 ~ 1MB bf16 fits the 192MB L2 easily).
__global__ __launch_bounds__(256)
void k_point_attn(const float* __restrict__ pos, const int* __restrict__ idx,
                  const float* __restrict__ qf, const float* __restrict__ kf,
                  const float* __restrict__ vvv,
                  const float* __restrict__ posw1,                 // [64 x 3] f32
                  const float* __restrict__ pos_scale, const float* __restrict__ pos_shift,
                  const bf16_t* __restrict__ posw2,                // [256 x 64]
                  const float* __restrict__ posb2,
                  const bf16_t* __restrict__ aw1,                  // [1024 x 256]
                  const float* __restrict__ a_scale, const float* __restrict__ a_shift,
                  const bf16_t* __restrict__ aw2,                  // [256 x 1024]
                  const float* __restrict__ ab2,
                  bf16_t* __restrict__ agg) {
  __shared__ bf16_t s_ain[16 * 256];    // attn-MLP input; reused for logits
  __shared__ bf16_t s_hid[16 * 1024];   // attn-MLP hidden
  __shared__ bf16_t s_pe[16 * 256];     // pos_emb
  __shared__ bf16_t s_ph[16 * 64];      // pos-MLP hidden
  __shared__ int s_idx[KNN];
  __shared__ float s_q[256];

  const int p = blockIdx.x;
  const int b = p / NPTS, n = p - b * NPTS;
  const int t = threadIdx.x;
  const int wave = t >> 5, lane = t & 31;
  const int m16 = lane & 15, half = lane >> 4;

  if (t < KNN) s_idx[t] = idx[(size_t)p * KNN + t];
  s_q[t] = qf[(size_t)p * DIM + t];
  __syncthreads();

  // pos-MLP layer 1: [16 x 3] -> [16 x 64], BN+ReLU (scalar, tiny)
  for (int e = t; e < KNN * POSH; e += 256) {
    int kk = e >> 6, h = e & 63;
    int q = s_idx[kk];
    float acc = 0.f;
#pragma unroll
    for (int d = 0; d < 3; ++d) {
      float rel = pos[(size_t)(b * 3 + d) * NPTS + n] - pos[(size_t)(b * 3 + d) * NPTS + q];
      acc += rel * posw1[h * 3 + d];
    }
    acc = acc * pos_scale[h] + pos_shift[h];
    s_ph[e] = (bf16_t)fmaxf(acc, 0.f);
  }
  __syncthreads();

  // pos-MLP layer 2 via WMMA: [16 x 64] @ [64 x 256]^T ; 16 tiles, 2/wave
  for (int tn = wave * 2; tn < wave * 2 + 2; ++tn) {
    v8f acc = {};
#pragma unroll
    for (int k0 = 0; k0 < POSH; k0 += 32) {
      v16bf a = load_frag(s_ph, POSH, m16, k0, half);
      v16bf w = load_frag(posw2, POSH, tn * 16 + m16, k0, half);
      acc = wmma_bf16(a, w, acc);
    }
    int nc = tn * 16 + m16;
    float bv = posb2[nc];
#pragma unroll
    for (int r = 0; r < 8; ++r)
      s_pe[(half * 8 + r) * DIM + nc] = (bf16_t)(acc[r] + bv);
  }
  __syncthreads();

  // attn-MLP input: qk_rel + pos_emb
  for (int e = t; e < KNN * DIM; e += 256) {
    int kk = e >> 8, c = e & 255;
    int q = s_idx[kk];
    float v = s_q[c] - kf[((size_t)(b * NPTS + q)) * DIM + c] + (float)s_pe[e];
    s_ain[e] = (bf16_t)v;
  }
  __syncthreads();

  // GEMM1: hidden = relu(bn(a_in @ aw1^T)) ; 64 n-tiles, 8/wave, K=256
  for (int tn = wave * 8; tn < wave * 8 + 8; ++tn) {
    v8f acc = {};
#pragma unroll
    for (int k0 = 0; k0 < DIM; k0 += 32) {
      v16bf a = load_frag(s_ain, DIM, m16, k0, half);
      v16bf w = load_frag(aw1, DIM, tn * 16 + m16, k0, half);
      acc = wmma_bf16(a, w, acc);
    }
    int nc = tn * 16 + m16;
    float sc = a_scale[nc], sh = a_shift[nc];
#pragma unroll
    for (int r = 0; r < 8; ++r) {
      float v = fmaxf(acc[r] * sc + sh, 0.f);
      s_hid[(half * 8 + r) * FFNH + nc] = (bf16_t)v;
    }
  }
  __syncthreads();

  // GEMM2: logits = hidden @ aw2^T + b2 ; 16 n-tiles, 2/wave, K=1024 -> s_ain
  for (int tn = wave * 2; tn < wave * 2 + 2; ++tn) {
    v8f acc = {};
    for (int k0 = 0; k0 < FFNH; k0 += 32) {
      v16bf a = load_frag(s_hid, FFNH, m16, k0, half);
      v16bf w = load_frag(aw2, FFNH, tn * 16 + m16, k0, half);
      acc = wmma_bf16(a, w, acc);
    }
    int nc = tn * 16 + m16;
    float bv = ab2[nc];
#pragma unroll
    for (int r = 0; r < 8; ++r)
      s_ain[(half * 8 + r) * DIM + nc] = (bf16_t)(acc[r] + bv);
  }
  __syncthreads();

  // softmax over the K=16 neighbors (rows) per channel; aggregate value_g
  {
    int c = t;
    float vals[KNN];
    float mx = -3.0e38f;
#pragma unroll
    for (int kk = 0; kk < KNN; ++kk) {
      vals[kk] = (float)s_ain[kk * DIM + c];
      mx = fmaxf(mx, vals[kk]);
    }
    float s = 0.f;
#pragma unroll
    for (int kk = 0; kk < KNN; ++kk) { vals[kk] = __expf(vals[kk] - mx); s += vals[kk]; }
    float inv = 1.f / s;
    float vc = vvv[(size_t)p * DIM + c];
    float a = 0.f;
#pragma unroll
    for (int kk = 0; kk < KNN; ++kk) {
      int q = s_idx[kk];
      float vg = vvv[((size_t)(b * NPTS + q)) * DIM + c] - vc + (float)s_pe[kk * DIM + c];
      a += vals[kk] * inv * vg;
    }
    agg[(size_t)p * DIM + c] = (bf16_t)a;
  }
}

// final: out(B,128,N) = conv_end + identity, from point-major [PTS x 128]
__global__ void k_final(const float* __restrict__ endv, const float* __restrict__ iden,
                        float* __restrict__ out) {
  int e = blockIdx.x * blockDim.x + threadIdx.x;
  if (e >= PTS * INCH) return;
  int p = e >> 7, c = e & 127;
  int b = p / NPTS, n = p - b * NPTS;
  out[((size_t)b * INCH + c) * NPTS + n] = endv[e] + iden[e];
}

// ---------------- host-side helpers ----------------
static inline void gemm_launch(hipStream_t s, const bf16_t* X, int ldx,
                               const bf16_t* W, const float* bias, const float* Res,
                               float* Out, bf16_t* OutBf, int I, int O, int epi) {
  dim3 grid(O / 16, PTS / 16);
  k_gemm<<<grid, 32, 0, s>>>(X, ldx, W, bias, Res, Out, OutBf, I, O, epi);
}

struct RwkvScratch {
  float *x0, *xn, *k32, *v32, *r32, *wkv, *x1, *xn2, *tmp, *r2;
  bf16_t *xk, *xv, *xr, *rw, *xk2, *xr2, *kk;
};

struct RwkvWeights { bf16_t *wk, *wv, *wr, *wo, *fwk, *fwv, *fwr; };

static void run_rwkv(hipStream_t s, void* const* din, int base,
                     const RwkvWeights& w, const float* points,
                     float* out_f, bf16_t* out_bf, RwkvScratch& sc) {
  auto L = [&](int o) { return (const float*)din[base + o]; };
  const int ne = PTS * DIM;
  const dim3 gb((ne + 255) / 256);
  k_layernorm<<<PTS, 32, 0, s>>>(points, L(RW_LN0_G), L(RW_LN0_B), sc.x0);
  k_layernorm<<<PTS, 32, 0, s>>>(sc.x0, L(RW_LN1_G), L(RW_LN1_B), sc.xn);
  k_mixshift<<<gb, 256, 0, s>>>(sc.xn, L(RW_MIX_K), L(RW_MIX_V), L(RW_MIX_R),
                                sc.xk, sc.xv, sc.xr);
  gemm_launch(s, sc.xk, DIM, w.wk, nullptr, nullptr, sc.k32, nullptr, DIM, DIM, 0);
  gemm_launch(s, sc.xv, DIM, w.wv, nullptr, nullptr, sc.v32, nullptr, DIM, DIM, 0);
  gemm_launch(s, sc.xr, DIM, w.wr, nullptr, nullptr, sc.r32, nullptr, DIM, DIM, 2);
  k_wkv<<<(BATCH * DIM + 255) / 256, 256, 0, s>>>(L(RW_TD), L(RW_TF), sc.k32, sc.v32, sc.wkv);
  k_mul_bf<<<gb, 256, 0, s>>>(sc.r32, sc.wkv, sc.rw, ne);
  gemm_launch(s, sc.rw, DIM, w.wo, nullptr, sc.x0, sc.x1, nullptr, DIM, DIM, 0);
  k_layernorm<<<PTS, 32, 0, s>>>(sc.x1, L(RW_LN2_G), L(RW_LN2_B), sc.xn2);
  k_mixshift<<<gb, 256, 0, s>>>(sc.xn2, L(RW_FMIX_K), nullptr, L(RW_FMIX_R),
                                sc.xk2, nullptr, sc.xr2);
  gemm_launch(s, sc.xk2, DIM, w.fwk, nullptr, nullptr, nullptr, sc.kk, DIM, FFNH, 3);
  gemm_launch(s, sc.kk, FFNH, w.fwv, nullptr, nullptr, sc.tmp, nullptr, FFNH, DIM, 0);
  gemm_launch(s, sc.xr2, DIM, w.fwr, nullptr, nullptr, sc.r2, nullptr, DIM, DIM, 2);
  k_fma_ew<<<gb, 256, 0, s>>>(sc.x1, sc.r2, sc.tmp, out_f, out_bf, ne);
}

extern "C" void kernel_launch(void* const* d_in, const int* in_sizes, int n_in,
                              void* d_out, int out_size, void* d_ws, size_t ws_size,
                              hipStream_t stream) {
  if (n_in < IN_TOTAL) return;  // unexpected flattening; bail deterministically
  (void)in_sizes; (void)ws_size;

  char* cur = (char*)d_ws;
  auto alloc = [&](size_t bytes) -> void* {
    void* p = (void*)cur;
    cur += (bytes + 255) & ~(size_t)255;
    return p;
  };
  auto fbuf = [&](size_t n) { return (float*)alloc(n * sizeof(float)); };
  auto bbuf = [&](size_t n) { return (bf16_t*)alloc(n * sizeof(bf16_t)); };
  auto castw = [&](int leaf, size_t n) -> bf16_t* {
    bf16_t* dst = bbuf(n);
    k_cast_bf<<<dim3((unsigned)((n + 255) / 256)), 256, 0, stream>>>(
        (const float*)d_in[leaf], dst, (int)n);
    return dst;
  };

  const float* pos   = (const float*)d_in[IN_POS];
  const float* key   = (const float*)d_in[IN_KEY];
  const float* query = (const float*)d_in[IN_QUERY];

  // ---- weight casts to bf16 ----
  bf16_t* c1w  = castw(IN_MLPV_C1_W, (size_t)INCH * 2 * INCH);
  bf16_t* c2w  = castw(IN_MLPV_C2_W, (size_t)INCH * INCH);
  bf16_t* scw  = castw(IN_MLPV_SC_W, (size_t)INCH * 2 * INCH);
  bf16_t* cvw  = castw(IN_CV_W,  (size_t)DIM * INCH);
  bf16_t* ckw  = castw(IN_CKEY_W, (size_t)DIM * INCH);
  bf16_t* cqw  = castw(IN_CQ_W,  (size_t)DIM * INCH);
  bf16_t* pw2  = castw(IN_PMLP_W2, (size_t)DIM * POSH);
  bf16_t* aw1  = castw(IN_AMLP_W1, (size_t)FFNH * DIM);
  bf16_t* aw2  = castw(IN_AMLP_W2, (size_t)DIM * FFNH);
  bf16_t* a1w  = castw(IN_ATTN1_W, (size_t)DIM * DIM);
  bf16_t* a2w  = castw(IN_ATTN2_W, (size_t)DIM * DIM);
  bf16_t* cew  = castw(IN_CEND_W, (size_t)INCH * DIM);
  RwkvWeights rw0 = {
    castw(IN_RWKV0 + RW_WK, (size_t)DIM * DIM), castw(IN_RWKV0 + RW_WV, (size_t)DIM * DIM),
    castw(IN_RWKV0 + RW_WR, (size_t)DIM * DIM), castw(IN_RWKV0 + RW_WO, (size_t)DIM * DIM),
    castw(IN_RWKV0 + RW_FWK, (size_t)FFNH * DIM), castw(IN_RWKV0 + RW_FWV, (size_t)DIM * FFNH),
    castw(IN_RWKV0 + RW_FWR, (size_t)DIM * DIM)};
  RwkvWeights rw1 = {
    castw(IN_RWKV1 + RW_WK, (size_t)DIM * DIM), castw(IN_RWKV1 + RW_WV, (size_t)DIM * DIM),
    castw(IN_RWKV1 + RW_WR, (size_t)DIM * DIM), castw(IN_RWKV1 + RW_WO, (size_t)DIM * DIM),
    castw(IN_RWKV1 + RW_FWK, (size_t)FFNH * DIM), castw(IN_RWKV1 + RW_FWV, (size_t)DIM * FFNH),
    castw(IN_RWKV1 + RW_FWR, (size_t)DIM * DIM)};

  // ---- activation buffers ----
  bf16_t* cat_bf   = bbuf((size_t)PTS * 2 * INCH);
  bf16_t* h_bf     = bbuf((size_t)PTS * INCH);
  float*  sc_f     = fbuf((size_t)PTS * INCH);
  float*  value_f  = fbuf((size_t)PTS * INCH);     // identity
  bf16_t* value_bf = bbuf((size_t)PTS * INCH);
  float*  vf_f     = fbuf((size_t)PTS * DIM);      // points (B,N,DIM)
  float*  kf_f     = fbuf((size_t)PTS * DIM);
  float*  qf_f     = fbuf((size_t)PTS * DIM);
  int*    idx      = (int*)alloc((size_t)PTS * KNN * sizeof(int));
  float*  a_f      = fbuf((size_t)PTS * DIM);
  bf16_t* a_bf     = bbuf((size_t)PTS * DIM);
  float*  v1_f     = fbuf((size_t)PTS * DIM);
  float*  asig_f   = fbuf((size_t)PTS * DIM);
  bf16_t* am_bf    = bbuf((size_t)PTS * DIM);
  float*  vvv_f    = fbuf((size_t)PTS * DIM);
  bf16_t* agg_bf   = bbuf((size_t)PTS * DIM);
  float*  end_f    = fbuf((size_t)PTS * INCH);
  float*  pscale   = fbuf(POSH);
  float*  pshift   = fbuf(POSH);
  float*  ascale   = fbuf(FFNH);
  float*  ashift   = fbuf(FFNH);
  RwkvScratch rsc;
  rsc.x0  = fbuf((size_t)PTS * DIM);  rsc.xn  = fbuf((size_t)PTS * DIM);
  rsc.k32 = fbuf((size_t)PTS * DIM);  rsc.v32 = fbuf((size_t)PTS * DIM);
  rsc.r32 = fbuf((size_t)PTS * DIM);  rsc.wkv = fbuf((size_t)PTS * DIM);
  rsc.x1  = fbuf((size_t)PTS * DIM);  rsc.xn2 = fbuf((size_t)PTS * DIM);
  rsc.tmp = fbuf((size_t)PTS * DIM);  rsc.r2  = fbuf((size_t)PTS * DIM);
  rsc.xk  = bbuf((size_t)PTS * DIM);  rsc.xv  = bbuf((size_t)PTS * DIM);
  rsc.xr  = bbuf((size_t)PTS * DIM);  rsc.rw  = bbuf((size_t)PTS * DIM);
  rsc.xk2 = bbuf((size_t)PTS * DIM);  rsc.xr2 = bbuf((size_t)PTS * DIM);
  rsc.kk  = bbuf((size_t)PTS * FFNH);

  // ---- 1) pack concat(key,query) point-major ----
  k_pack_cat<<<(PTS * 2 * INCH + 255) / 256, 256, 0, stream>>>(key, query, cat_bf);

  // ---- 2) mlp_v: value = c2(relu(c1(cat))) + sc(cat) ----
  gemm_launch(stream, cat_bf, 2 * INCH, c1w, (const float*)d_in[IN_MLPV_C1_B],
              nullptr, nullptr, h_bf, 2 * INCH, INCH, 1);
  gemm_launch(stream, cat_bf, 2 * INCH, scw, (const float*)d_in[IN_MLPV_SC_B],
              nullptr, sc_f, nullptr, 2 * INCH, INCH, 0);
  gemm_launch(stream, h_bf, INCH, c2w, (const float*)d_in[IN_MLPV_C2_B],
              sc_f, value_f, value_bf, INCH, INCH, 0);

  // ---- 3) 1x1 convs: value/key/query features ----
  gemm_launch(stream, value_bf, INCH, cvw, (const float*)d_in[IN_CV_B],
              nullptr, vf_f, nullptr, INCH, DIM, 0);
  gemm_launch(stream, cat_bf, 2 * INCH, ckw, (const float*)d_in[IN_CKEY_B],
              nullptr, kf_f, nullptr, INCH, DIM, 0);
  gemm_launch(stream, cat_bf + INCH, 2 * INCH, cqw, (const float*)d_in[IN_CQ_B],
              nullptr, qf_f, nullptr, INCH, DIM, 0);

  // ---- 4) kNN ----
  k_knn<<<(PTS + 255) / 256, 256, 0, stream>>>(pos, idx);

  // ---- 5) two RWKV blocks on points ----
  run_rwkv(stream, d_in, IN_RWKV0, rw0, vf_f, a_f, a_bf, rsc);
  run_rwkv(stream, d_in, IN_RWKV1, rw1, vf_f, v1_f, nullptr, rsc);

  // ---- 6) vvv = (sigmoid(a@attn1^T) * v1) @ attn2^T ----
  gemm_launch(stream, a_bf, DIM, a1w, nullptr, nullptr, asig_f, nullptr, DIM, DIM, 2);
  k_mul_bf<<<(PTS * DIM + 255) / 256, 256, 0, stream>>>(asig_f, v1_f, am_bf, PTS * DIM);
  gemm_launch(stream, am_bf, DIM, a2w, nullptr, nullptr, vvv_f, nullptr, DIM, DIM, 0);

  // ---- 7) BN prefold (scale/shift absorb conv bias) ----
  k_bnprep<<<1, 64, 0, stream>>>((const float*)d_in[IN_PMLP_BN_G], (const float*)d_in[IN_PMLP_BN_B],
                                 (const float*)d_in[IN_PMLP_BN_M], (const float*)d_in[IN_PMLP_BN_V],
                                 (const float*)d_in[IN_PMLP_B1], pscale, pshift, POSH);
  k_bnprep<<<4, 256, 0, stream>>>((const float*)d_in[IN_AMLP_BN_G], (const float*)d_in[IN_AMLP_BN_B],
                                  (const float*)d_in[IN_AMLP_BN_M], (const float*)d_in[IN_AMLP_BN_V],
                                  (const float*)d_in[IN_AMLP_B1], ascale, ashift, FFNH);

  // ---- 8) fused per-point neighbor attention ----
  k_point_attn<<<PTS, 256, 0, stream>>>(pos, idx, qf_f, kf_f, vvv_f,
                                        (const float*)d_in[IN_PMLP_W1], pscale, pshift,
                                        pw2, (const float*)d_in[IN_PMLP_B2],
                                        aw1, ascale, ashift,
                                        aw2, (const float*)d_in[IN_AMLP_B2], agg_bf);

  // ---- 9) conv_end + identity -> (B,128,N) ----
  gemm_launch(stream, agg_bf, DIM, cew, (const float*)d_in[IN_CEND_B],
              nullptr, end_f, nullptr, DIM, INCH, 0);
  k_final<<<(PTS * INCH + 255) / 256, 256, 0, stream>>>(end_f, value_f, (float*)d_out);
}
